// Corrector_51591147159911
// MI455X (gfx1250) — matrix operand order repaired and checked by reading, not compiled, not run on hardware
//
#include <hip/hip_runtime.h>

typedef _Float16 half4  __attribute__((ext_vector_type(4)));
typedef _Float16 half8  __attribute__((ext_vector_type(8)));
typedef _Float16 half16 __attribute__((ext_vector_type(16)));
typedef float    v8f    __attribute__((ext_vector_type(8)));

#define DD     256
#define NTOK   4096
#define BATCH  32
#define QSCALE 0.0625f   // 256^-0.5

// ---------------------------------------------------------------- helpers

__device__ __forceinline__ v8f wmma16(half16 a, half16 b, v8f c) {
  // D = A(16x32 f16) * B(32x16 f16) + C(16x16 f32)
  return __builtin_amdgcn_wmma_f32_16x16x32_f16(false, a, false, b, (short)0, c,
                                                false, false);
}

__device__ __forceinline__ half16 ld2x8(const _Float16* p0, const _Float16* p1) {
  half16 r;
  ((half8*)&r)[0] = *(const half8*)p0;
  ((half8*)&r)[1] = *(const half8*)p1;
  return r;
}

__device__ __forceinline__ v8f zero8() {
  v8f z = {0.f, 0.f, 0.f, 0.f, 0.f, 0.f, 0.f, 0.f};
  return z;
}

// A operand (16x32 f16): interleaved quarter layout per ISA A-table.
//   lanes 0-15 : K in {kk*32+0..7} U {kk*32+16..23}
//   lanes 16-31: K in {kk*32+8..15} U {kk*32+24..31}
__device__ __forceinline__ half16 ldA(const _Float16* rowbase, int kk, int hf) {
  const _Float16* p = rowbase + kk * 32 + hf * 8;
  return ld2x8(p, p + 16);
}

// B operand (32x16 f16): interleaved quarter layout matching A.
//   lanes 0-15 : K in {kk*32+0..7} U {kk*32+16..23}
//   lanes 16-31: K in {kk*32+8..15} U {kk*32+24..31}
__device__ __forceinline__ half16 ldB(const _Float16* colbase, int kk, int hf) {
  const _Float16* p = colbase + kk * 32 + hf * 8;
  return ld2x8(p, p + 16);
}

// ---------------------------------------------------------------- utility kernels

__global__ __launch_bounds__(256) void cvt_f32_f16_kernel(const float* __restrict__ src,
                                                          _Float16* __restrict__ dst,
                                                          int n) {
  int i = blockIdx.x * 256 + threadIdx.x;
  if (i < n) dst[i] = (_Float16)src[i];
}

__global__ __launch_bounds__(256) void zero_f32_kernel(float* __restrict__ dst, int n) {
  int i = blockIdx.x * 256 + threadIdx.x;
  if (i < n) dst[i] = 0.f;
}

// ---------------------------------------------------------------- LN(inputs) + K/V GEMM
// grid.x = (B*N)/(8 waves * 16 rows) = 1024, block = 256 (8 waves)
__global__ __launch_bounds__(256) void ln_kv_kernel(
    const float* __restrict__ x, const float* __restrict__ g, const float* __restrict__ bt,
    const _Float16* __restrict__ Wk16, const float* __restrict__ bk,
    const _Float16* __restrict__ Wv16, const float* __restrict__ bv,
    _Float16* __restrict__ k16, _Float16* __restrict__ v16) {
  __shared__ __align__(16) _Float16 xln[8][16 * DD];   // 64 KB: per-wave A tiles

  const int wave = threadIdx.x >> 5, lane = threadIdx.x & 31;
  const int m = lane & 15, hf = lane >> 4;
  const long rowBase = ((long)blockIdx.x * 8 + wave) * 16;
  const long row = rowBase + m;
  const float* xr = x + row * DD + hf * 128;

  // pass 1: per-row mean/var (lane pair l, l^16 split the 256 cols)
  float s = 0.f, sq = 0.f;
  for (int i = 0; i < 32; ++i) {
    float4 t = ((const float4*)xr)[i];
    s  += t.x + t.y + t.z + t.w;
    sq += t.x * t.x + t.y * t.y + t.z * t.z + t.w * t.w;
  }
  s  += __shfl_xor(s, 16, 32);
  sq += __shfl_xor(sq, 16, 32);
  const float mean = s * (1.f / DD);
  const float rstd = rsqrtf(sq * (1.f / DD) - mean * mean + 1e-3f);

  // pass 2: normalize -> f16 A tile in LDS (row-major [m][c])
  _Float16* xw = &xln[wave][0];
  for (int i = 0; i < 32; ++i) {
    const int c = hf * 128 + i * 4;
    float4 t  = ((const float4*)xr)[i];
    float4 gg = *(const float4*)(g + c);
    float4 bb = *(const float4*)(bt + c);
    half4 hv;
    hv.x = (_Float16)((t.x - mean) * rstd * gg.x + bb.x);
    hv.y = (_Float16)((t.y - mean) * rstd * gg.y + bb.y);
    hv.z = (_Float16)((t.z - mean) * rstd * gg.z + bb.z);
    hv.w = (_Float16)((t.w - mean) * rstd * gg.w + bb.w);
    *(half4*)(xw + m * DD + c) = hv;
  }
  __syncthreads();

  const _Float16* arow = xw + m * DD;
  for (int wsel = 0; wsel < 2; ++wsel) {
    const _Float16* W    = wsel ? Wv16 : Wk16;
    const float*    bias = wsel ? bv : bk;
    _Float16*       out  = wsel ? v16 : k16;
    for (int nc = 0; nc < 4; ++nc) {
      v8f acc[4];
      #pragma unroll
      for (int t = 0; t < 4; ++t) acc[t] = zero8();
      for (int kk = 0; kk < 8; ++kk) {
        half16 a = ldA(arow, kk, hf);
        #pragma unroll
        for (int t = 0; t < 4; ++t) {
          const int o = (nc * 4 + t) * 16 + m;               // out column
          half16 bm = ldB(W + (long)o * DD, kk, hf);
          acc[t] = wmma16(a, bm, acc[t]);
        }
      }
      #pragma unroll
      for (int t = 0; t < 4; ++t) {
        const int o = (nc * 4 + t) * 16 + m;
        const float bo = bias[o];
        #pragma unroll
        for (int r = 0; r < 8; ++r) {
          const long rr = rowBase + r + hf * 8;
          out[rr * DD + o] = (_Float16)(acc[t][r] + bo);
        }
      }
    }
  }
}

// ---------------------------------------------------------------- LN(slots) + q projection
// grid = B, block = 256. q16 padded to 16 rows (rows 8..15 zero).
__global__ __launch_bounds__(256) void slot_q_kernel(
    const float* __restrict__ slots, const float* __restrict__ g, const float* __restrict__ bb,
    const float* __restrict__ Wq, const float* __restrict__ bq,
    _Float16* __restrict__ q16) {
  __shared__ float sn[8][DD];
  const int b = blockIdx.x;
  const int wave = threadIdx.x >> 5, lane = threadIdx.x & 31;

  const float* hr = slots + ((long)b * 8 + wave) * DD;
  float vals[8];
  float s = 0.f, sq = 0.f;
  #pragma unroll
  for (int e = 0; e < 8; ++e) {
    float t = hr[lane * 8 + e];
    vals[e] = t; s += t; sq += t * t;
  }
  for (int off = 16; off >= 1; off >>= 1) {
    s += __shfl_xor(s, off, 32); sq += __shfl_xor(sq, off, 32);
  }
  const float mean = s / 256.f;
  const float rstd = rsqrtf(sq / 256.f - mean * mean + 1e-3f);
  #pragma unroll
  for (int e = 0; e < 8; ++e) {
    int c = lane * 8 + e;
    sn[wave][c] = (vals[e] - mean) * rstd * g[c] + bb[c];
  }
  __syncthreads();

  const int i = threadIdx.x >> 5;      // slot row
  const int ob = threadIdx.x & 31;
  for (int u = 0; u < 8; ++u) {
    const int o = ob + u * 32;
    const float* wr = Wq + (long)o * DD;
    float acc = bq[o];
    for (int j = 0; j < DD; ++j) acc += sn[i][j] * wr[j];
    q16[((long)b * 16 + i) * DD + o]     = (_Float16)acc;
    q16[((long)b * 16 + 8 + i) * DD + o] = (_Float16)0.f;   // padding rows
  }
}

// ---------------------------------------------------------------- dots + softmax-over-slots
// grid = (32 token-blocks, B), block = 256 (8 waves, 16 tokens each)
__global__ __launch_bounds__(256) void dots_softmax_kernel(
    const _Float16* __restrict__ q16, const _Float16* __restrict__ k16,
    _Float16* __restrict__ p16, float* __restrict__ rowsum) {
  __shared__ __align__(16) _Float16 qs[16 * DD];   // 8 KB, shared A
  const int b = blockIdx.y;
  {
    const uint4* src = (const uint4*)(q16 + (long)b * 16 * DD);
    uint4* dst = (uint4*)qs;
    for (int i = threadIdx.x; i < 16 * DD / 8; i += 256) dst[i] = src[i];
  }
  __syncthreads();

  const int wave = threadIdx.x >> 5, lane = threadIdx.x & 31;
  const int m = lane & 15, hf = lane >> 4;
  const int tok = blockIdx.x * 128 + wave * 16 + m;

  v8f acc = zero8();
  const _Float16* arow = qs + m * DD;
  const _Float16* kb = k16 + ((long)b * NTOK + tok) * DD;   // B column = this token's k row
  for (int kk = 0; kk < 8; ++kk) {
    half16 a  = ldA(arow, kk, hf);
    half16 bm = ldB(kb, kk, hf);
    acc = wmma16(a, bm, acc);
  }

  // lanes 0..15 hold slots 0..7 for this token in acc[0..7]
  float p[8];
  float mx = -1e30f;
  #pragma unroll
  for (int i = 0; i < 8; ++i) { p[i] = acc[i] * QSCALE; mx = fmaxf(mx, p[i]); }
  float sum = 0.f;
  #pragma unroll
  for (int i = 0; i < 8; ++i) { p[i] = __expf(p[i] - mx); sum += p[i]; }
  const float inv = 1.f / sum;
  #pragma unroll
  for (int i = 0; i < 8; ++i) p[i] = p[i] * inv + 1e-8f;

  if (lane < 16) {
    #pragma unroll
    for (int i = 0; i < 8; ++i)
      p16[((long)b * 16 + i) * NTOK + tok] = (_Float16)p[i];
  } else {
    #pragma unroll
    for (int i = 0; i < 8; ++i)
      p16[((long)b * 16 + 8 + i) * NTOK + tok] = (_Float16)0.f;  // padding rows
  }

  // partial token-axis sums for the renormalization
  #pragma unroll
  for (int i = 0; i < 8; ++i) {
    float pv = (lane < 16) ? p[i] : 0.f;
    for (int off = 16; off >= 1; off >>= 1) pv += __shfl_xor(pv, off, 32);
    if (lane == 0) atomicAdd(rowsum + b * 8 + i, pv);
  }
}

// ---------------------------------------------------------------- updates = (p/rowsum) @ v
// grid = (8 k-splits, B), block = 256 (8 waves x 2 n-tiles = full D)
__global__ __launch_bounds__(256) void updates_kernel(
    const _Float16* __restrict__ p16, const _Float16* __restrict__ v16,
    const float* __restrict__ rowsum, float* __restrict__ upd) {
  __shared__ __align__(16) _Float16 vt[256 * 48];   // 24 KB, [d][tok], stride 48 keeps 16B ds alignment
  const int b = blockIdx.y;
  const int wave = threadIdx.x >> 5, lane = threadIdx.x & 31;
  const int m = lane & 15, hf = lane >> 4;
  const int d0 = wave * 32 + m;
  const int d1 = wave * 32 + 16 + m;

  v8f acc0 = zero8(), acc1 = zero8();
  for (int kt = 0; kt < 16; ++kt) {
    const int tokbase = (blockIdx.x * 16 + kt) * 32;
    __syncthreads();
    {   // cooperative transpose of v[b, tokbase..+31, :] into LDS [d][tok]
      const int r = threadIdx.x >> 3;           // 0..31 token in tile
      const int cbase = (threadIdx.x & 7) * 32; // 0..224 column base
      const _Float16* vr = v16 + ((long)b * NTOK + tokbase + r) * DD + cbase;
      __builtin_prefetch(vr + 32 * DD, 0, 1);   // next k-step tile -> global_prefetch_b8
      #pragma unroll
      for (int u = 0; u < 4; ++u) {
        half8 hv = *(const half8*)(vr + u * 8);
        #pragma unroll
        for (int e = 0; e < 8; ++e) vt[(cbase + u * 8 + e) * 48 + r] = hv[e];
      }
    }
    __syncthreads();

    const _Float16* ap = p16 + ((long)b * 16 + m) * NTOK + tokbase;
    half16 a = ldA(ap, 0, hf);
    half16 bm0 = ldB(vt + d0 * 48, 0, hf);
    acc0 = wmma16(a, bm0, acc0);
    half16 bm1 = ldB(vt + d1 * 48, 0, hf);
    acc1 = wmma16(a, bm1, acc1);
  }

  const int mrow0 = hf * 8;
  #pragma unroll
  for (int r = 0; r < 8; ++r) {
    const int mm = r + mrow0;
    if (mm < 8) {   // rows 8..15 are padding
      const float sc = 1.f / rowsum[b * 8 + mm];
      atomicAdd(upd + ((long)b * 8 + mm) * DD + d0, acc0[r] * sc);
      atomicAdd(upd + ((long)b * 8 + mm) * DD + d1, acc1[r] * sc);
    }
  }
}

// ---------------------------------------------------------------- GRU cell + MLP (+residual)
// grid = B, block = 256
__global__ __launch_bounds__(256) void gru_mlp_kernel(
    const float* __restrict__ upd, const float* __restrict__ slots_src,
    const float* __restrict__ W_ih, const float* __restrict__ b_ih,
    const float* __restrict__ W_hh, const float* __restrict__ b_hh,
    const float* __restrict__ lng, const float* __restrict__ lnb,
    const float* __restrict__ W1, const float* __restrict__ b1,
    const float* __restrict__ W2, const float* __restrict__ b2,
    float* __restrict__ slots_dst, float* __restrict__ outp) {
  __shared__ float u[8][DD], h[8][DD], hn[8][DD], tln[8][DD], act[8][512];
  const int b = blockIdx.x;
  for (int i = threadIdx.x; i < 8 * DD; i += 256) {
    (&u[0][0])[i] = upd[(long)b * 8 * DD + i];
    (&h[0][0])[i] = slots_src[(long)b * 8 * DD + i];
  }
  __syncthreads();

  const int row = threadIdx.x >> 5, cb = threadIdx.x & 31;
  for (int uu = 0; uu < 8; ++uu) {
    const int c = cb + uu * 32;
    float ir = b_ih[c],        hr  = b_hh[c];
    float iz = b_ih[256 + c],  hz  = b_hh[256 + c];
    float in_ = b_ih[512 + c], hnn = b_hh[512 + c];
    const float* wir = W_ih + (long)c * DD;
    const float* wiz = W_ih + (long)(256 + c) * DD;
    const float* win = W_ih + (long)(512 + c) * DD;
    const float* whr = W_hh + (long)c * DD;
    const float* whz = W_hh + (long)(256 + c) * DD;
    const float* whn = W_hh + (long)(512 + c) * DD;
    for (int j = 0; j < DD; ++j) {
      const float uj = u[row][j], hj = h[row][j];
      ir  += uj * wir[j];  iz  += uj * wiz[j];  in_ += uj * win[j];
      hr  += hj * whr[j];  hz  += hj * whz[j];  hnn += hj * whn[j];
    }
    const float r = 1.f / (1.f + __expf(-(ir + hr)));
    const float z = 1.f / (1.f + __expf(-(iz + hz)));
    const float n = tanhf(in_ + r * hnn);
    hn[row][c] = (1.f - z) * n + z * h[row][c];
  }
  __syncthreads();

  {   // LayerNorm(hn): wave per row
    const int w = threadIdx.x >> 5, lane = threadIdx.x & 31;
    float s = 0.f, sq = 0.f;
    #pragma unroll
    for (int e = 0; e < 8; ++e) { float t = hn[w][lane * 8 + e]; s += t; sq += t * t; }
    for (int off = 16; off >= 1; off >>= 1) {
      s += __shfl_xor(s, off, 32); sq += __shfl_xor(sq, off, 32);
    }
    const float mean = s / 256.f;
    const float rstd = rsqrtf(sq / 256.f - mean * mean + 1e-3f);
    #pragma unroll
    for (int e = 0; e < 8; ++e) {
      const int c = lane * 8 + e;
      tln[w][c] = (hn[w][c] - mean) * rstd * lng[c] + lnb[c];
    }
  }
  __syncthreads();

  for (int uu = 0; uu < 16; ++uu) {
    const int o = cb + uu * 32;
    const float* wr = W1 + (long)o * DD;
    float a = b1[o];
    for (int j = 0; j < DD; ++j) a += tln[row][j] * wr[j];
    act[row][o] = fmaxf(a, 0.f);
  }
  __syncthreads();

  for (int uu = 0; uu < 8; ++uu) {
    const int c = cb + uu * 32;
    const float* wr = W2 + (long)c * 512;
    float a = b2[c];
    for (int j = 0; j < 512; ++j) a += act[row][j] * wr[j];
    const float val = hn[row][c] + a;
    slots_dst[((long)b * 8 + row) * DD + c] = val;
    outp[((long)b * 8 + row) * DD + c]      = val;
  }
}

// ---------------------------------------------------------------- host launcher

extern "C" void kernel_launch(void* const* d_in, const int* in_sizes, int n_in,
                              void* d_out, int out_size, void* d_ws, size_t ws_size,
                              hipStream_t stream) {
  (void)in_sizes; (void)n_in; (void)out_size; (void)ws_size;
  const float* inputs   = (const float*)d_in[0];
  const float* slots_in = (const float*)d_in[1];
  // d_in[2] = step (unused: step==0 -> 3 iterations)
  const float* ln_in_g   = (const float*)d_in[3];
  const float* ln_in_b   = (const float*)d_in[4];
  const float* ln_slot_g = (const float*)d_in[5];
  const float* ln_slot_b = (const float*)d_in[6];
  const float* ln_mlp_g  = (const float*)d_in[7];
  const float* ln_mlp_b  = (const float*)d_in[8];
  const float* Wq = (const float*)d_in[9];
  const float* bq = (const float*)d_in[10];
  const float* Wk = (const float*)d_in[11];
  const float* bk = (const float*)d_in[12];
  const float* Wv = (const float*)d_in[13];
  const float* bv = (const float*)d_in[14];
  const float* W_ih = (const float*)d_in[15];
  const float* b_ih = (const float*)d_in[16];
  const float* W_hh = (const float*)d_in[17];
  const float* b_hh = (const float*)d_in[18];
  const float* W1 = (const float*)d_in[19];
  const float* b1 = (const float*)d_in[20];
  const float* W2 = (const float*)d_in[21];
  const float* b2 = (const float*)d_in[22];
  float* out = (float*)d_out;

  char* p = (char*)d_ws;
  auto alloc = [&](size_t bytes) { void* r = p; p += (bytes + 255) & ~(size_t)255; return r; };
  _Float16* k16  = (_Float16*)alloc((size_t)BATCH * NTOK * DD * 2);   // 64 MB
  _Float16* v16  = (_Float16*)alloc((size_t)BATCH * NTOK * DD * 2);   // 64 MB
  _Float16* q16  = (_Float16*)alloc((size_t)BATCH * 16 * DD * 2);
  _Float16* p16  = (_Float16*)alloc((size_t)BATCH * 16 * NTOK * 2);   // 4 MB
  float*    rsum = (float*)alloc((size_t)BATCH * 8 * 4);
  float*    upd  = (float*)alloc((size_t)BATCH * 8 * DD * 4);
  float*    slws = (float*)alloc((size_t)BATCH * 8 * DD * 4);
  _Float16* wk16 = (_Float16*)alloc((size_t)DD * DD * 2);
  _Float16* wv16 = (_Float16*)alloc((size_t)DD * DD * 2);

  const int nw = DD * DD;   // 65536
  cvt_f32_f16_kernel<<<(nw + 255) / 256, 256, 0, stream>>>(Wk, wk16, nw);
  cvt_f32_f16_kernel<<<(nw + 255) / 256, 256, 0, stream>>>(Wv, wv16, nw);

  ln_kv_kernel<<<1024, 256, 0, stream>>>(inputs, ln_in_g, ln_in_b,
                                         wk16, bk, wv16, bv, k16, v16);

  for (int it = 0; it < 3; ++it) {
    const float* ssrc = (it == 0) ? slots_in : slws;
    zero_f32_kernel<<<1, 256, 0, stream>>>(rsum, BATCH * 8);
    zero_f32_kernel<<<(BATCH * 8 * DD + 255) / 256, 256, 0, stream>>>(upd, BATCH * 8 * DD);
    slot_q_kernel<<<BATCH, 256, 0, stream>>>(ssrc, ln_slot_g, ln_slot_b, Wq, bq, q16);
    dots_softmax_kernel<<<dim3(NTOK / 128, BATCH), 256, 0, stream>>>(q16, k16, p16, rsum);
    updates_kernel<<<dim3(8, BATCH), 256, 0, stream>>>(p16, v16, rsum, upd);
    gru_mlp_kernel<<<BATCH, 256, 0, stream>>>(upd, ssrc, W_ih, b_ih, W_hh, b_hh,
                                              ln_mlp_g, ln_mlp_b, W1, b1, W2, b2,
                                              slws, out);
  }
}
